// myGraph_gcn_25074019074259
// MI455X (gfx1250) — compile-verified, matrix operand order checked by reading, not run on hardware
//
#include <hip/hip_runtime.h>
#include <hip/hip_bf16.h>

#define FEAT   128
#define OUTF   8
#define NGRAPH 512

typedef __attribute__((ext_vector_type(16))) __bf16 v16bf;
typedef __attribute__((ext_vector_type(8)))  float  v8f;

union ABfrag { v16bf v; unsigned int u[8]; };

__device__ __forceinline__ unsigned int pack_bf2(float lo, float hi) {
    unsigned int a = __float_as_uint(lo);
    unsigned int b = __float_as_uint(hi);
    a = (a + 0x7FFFu + ((a >> 16) & 1u)) >> 16;   // round-to-nearest-even bf16
    b = (b + 0x7FFFu + ((b >> 16) & 1u)) >> 16;
    return (a & 0xFFFFu) | ((b & 0xFFFFu) << 16);
}

// ---------------------------------------------------------------------------
// H[nrows,128] = (relu_in ? relu(X) : X) @ W   (fp32 in, bf16 WMMA, fp32 out)
// 256 threads = 8 waves; each wave computes a 16x128 strip via 8 WMMA tiles.
// ---------------------------------------------------------------------------
__global__ __launch_bounds__(256)
void gcn_gemm_wmma(const float* __restrict__ X, const float* __restrict__ W,
                   float* __restrict__ H, int nrows, int relu_in) {
    __shared__ unsigned int wlds[64 * FEAT];   // bf16 pairs along K: [kk][n]

    const int tid = threadIdx.x;
    // Stage W (128x128 f32) into LDS as packed bf16 (K=2kk, 2kk+1) pairs.
    for (int i = tid; i < 64 * FEAT; i += 256) {
        int kk = i >> 7, n = i & 127;
        wlds[i] = pack_bf2(W[(2 * kk) * FEAT + n], W[(2 * kk + 1) * FEAT + n]);
    }
    __syncthreads();

    const int wave = tid >> 5;
    const int lane = tid & 31;
    const int half = lane >> 4;    // K-group select per ISA A/B bf16 layout
    const int l16  = lane & 15;

    int row  = blockIdx.x * 128 + wave * 16 + l16;
    int rowc = row < nrows ? row : nrows - 1;          // clamp loads, keep EXEC full
    const float* xrow = X + (size_t)rowc * FEAT;

    v8f acc[8];
#pragma unroll
    for (int t = 0; t < 8; ++t) {
        v8f z = {0.f, 0.f, 0.f, 0.f, 0.f, 0.f, 0.f, 0.f};
        acc[t] = z;
    }

    for (int k0 = 0; k0 < FEAT; k0 += 32) {
        // ----- A fragment: lane holds row, K = {c0..c0+7, c0+16..c0+23} -----
        ABfrag a;
        const int c0 = k0 + half * 8;
        const float4 f0 = *(const float4*)(xrow + c0);
        const float4 f1 = *(const float4*)(xrow + c0 + 4);
        const float4 f2 = *(const float4*)(xrow + c0 + 16);
        const float4 f3 = *(const float4*)(xrow + c0 + 20);
        float va[16] = {f0.x, f0.y, f0.z, f0.w, f1.x, f1.y, f1.z, f1.w,
                        f2.x, f2.y, f2.z, f2.w, f3.x, f3.y, f3.z, f3.w};
        if (relu_in) {
#pragma unroll
            for (int j = 0; j < 16; ++j) va[j] = va[j] > 0.f ? va[j] : 0.f;
        }
#pragma unroll
        for (int j = 0; j < 8; ++j) a.u[j] = pack_bf2(va[2 * j], va[2 * j + 1]);

        const int kkbase = (k0 >> 1) + half * 4;
#pragma unroll
        for (int t = 0; t < 8; ++t) {
            // ----- B fragment from LDS: lane holds col n, same K-groups -----
            ABfrag b;
            const int n = t * 16 + l16;
#pragma unroll
            for (int j = 0; j < 4; ++j) {
                b.u[j]     = wlds[(kkbase + j) * FEAT + n];
                b.u[4 + j] = wlds[(kkbase + 8 + j) * FEAT + n];
            }
            acc[t] = __builtin_amdgcn_wmma_f32_16x16x32_bf16(
                false, a.v, false, b.v, (short)0, acc[t], false, false);
        }
    }

    // C/D layout: VGPR r -> M = r (+8 for lanes 16..31), N = lane&15
    const int rbase = blockIdx.x * 128 + wave * 16 + half * 8;
#pragma unroll
    for (int r = 0; r < 8; ++r) {
        const int orow = rbase + r;
        if (orow < nrows) {
            float* hrow = H + (size_t)orow * FEAT;
#pragma unroll
            for (int t = 0; t < 8; ++t) hrow[t * 16 + l16] = acc[t][r];
        }
    }
}

// ---------------------------------------------------------------------------
// Small helper kernels
// ---------------------------------------------------------------------------
__global__ void zero_f32(float* p, int n) {
    int i = blockIdx.x * blockDim.x + threadIdx.x;
    if (i < n) p[i] = 0.f;
}

__global__ void deg_edges(const int* __restrict__ dst, float* __restrict__ deg, int ne) {
    int e = blockIdx.x * blockDim.x + threadIdx.x;
    if (e < ne) atomicAdd(&deg[dst[e]], 1.0f);
}

__global__ void make_dinv(const float* __restrict__ deg, float* __restrict__ dinv, int n) {
    int i = blockIdx.x * blockDim.x + threadIdx.x;
    if (i < n) dinv[i] = rsqrtf(deg[i] + 1.0f);   // +1 for self-loop
}

__global__ void make_norm_e(const int* __restrict__ src, const int* __restrict__ dst,
                            const float* __restrict__ dinv, float* __restrict__ ne,
                            int nedges) {
    int e = blockIdx.x * blockDim.x + threadIdx.x;
    if (e < nedges) ne[e] = dinv[src[e]] * dinv[dst[e]];
}

// out[i,f] = h[i,f] * dinv[i]^2 + bias[f]   (self-loop + bias init)
__global__ void agg_init(const float* __restrict__ h, const float* __restrict__ dinv,
                         const float* __restrict__ bias, float* __restrict__ out, int n) {
    int t = blockIdx.x * blockDim.x + threadIdx.x;
    if (t < n * FEAT) {
        int i = t >> 7, f = t & 127;
        float di = dinv[i];
        out[t] = h[t] * di * di + bias[f];
    }
}

// out[dst[e]] += h[src[e]] * norm_e[e]; wave-coherent e, float4 per thread
__global__ void edge_scatter(const float* __restrict__ h, const int* __restrict__ src,
                             const int* __restrict__ dst, const float* __restrict__ ne,
                             float* __restrict__ out, int nedges) {
    int t = blockIdx.x * blockDim.x + threadIdx.x;
    if (t < nedges * 32) {
        int e = t >> 5, f = (t & 31) * 4;
        int s = src[e], d = dst[e];
        float w = ne[e];
        const float4 v = *(const float4*)(h + (size_t)s * FEAT + f);
        float* o = out + (size_t)d * FEAT + f;
        atomicAdd(o + 0, v.x * w);
        atomicAdd(o + 1, v.y * w);
        atomicAdd(o + 2, v.z * w);
        atomicAdd(o + 3, v.w * w);
    }
}

__global__ void pool_sum(const float* __restrict__ h, const int* __restrict__ batch,
                         float* __restrict__ sums, int n) {
    int t = blockIdx.x * blockDim.x + threadIdx.x;
    if (t < n * FEAT) {
        int i = t >> 7, f = t & 127;
        atomicAdd(&sums[(size_t)batch[i] * FEAT + f], h[t]);
    }
}

__global__ void pool_cnt(const int* __restrict__ batch, float* __restrict__ counts, int n) {
    int i = blockIdx.x * blockDim.x + threadIdx.x;
    if (i < n) atomicAdd(&counts[batch[i]], 1.0f);
}

// logits[g,o] = (sums[g,:]/max(cnt,1)) @ Wlin[:,o] + blin[o]   (512x8, tiny)
__global__ void head_linear(const float* __restrict__ sums, const float* __restrict__ counts,
                            const float* __restrict__ Wlin, const float* __restrict__ blin,
                            float* __restrict__ out) {
    int t = blockIdx.x * blockDim.x + threadIdx.x;
    if (t < NGRAPH * OUTF) {
        int g = t >> 3, o = t & 7;
        float inv = 1.0f / fmaxf(counts[g], 1.0f);
        float acc = blin[o];
        for (int k = 0; k < FEAT; ++k)
            acc += sums[g * FEAT + k] * inv * Wlin[k * OUTF + o];
        out[t] = acc;
    }
}

// ---------------------------------------------------------------------------
extern "C" void kernel_launch(void* const* d_in, const int* in_sizes, int n_in,
                              void* d_out, int out_size, void* d_ws, size_t ws_size,
                              hipStream_t stream) {
    const float* x    = (const float*)d_in[0];
    const int*   ei   = (const int*)d_in[1];           // [2, E]
    const int*   bat  = (const int*)d_in[3];
    const float* W1   = (const float*)d_in[4];
    const float* b1   = (const float*)d_in[5];
    const float* W2   = (const float*)d_in[6];
    const float* b2   = (const float*)d_in[7];
    const float* W3   = (const float*)d_in[8];
    const float* b3   = (const float*)d_in[9];
    const float* Wlin = (const float*)d_in[10];
    const float* blin = (const float*)d_in[11];
    float*       out  = (float*)d_out;

    const int N = in_sizes[0] / FEAT;
    const int E = in_sizes[1] / 2;
    const int* src = ei;
    const int* dst = ei + E;

    // Workspace carve-up (all float)
    float* deg    = (float*)d_ws;                 // N
    float* dinv   = deg + N;                      // N
    float* norme  = dinv + N;                     // E
    float* bufA   = norme + E;                    // N*FEAT
    float* bufB   = bufA + (size_t)N * FEAT;      // N*FEAT
    float* sums   = bufB + (size_t)N * FEAT;      // NGRAPH*FEAT
    float* counts = sums + NGRAPH * FEAT;         // NGRAPH

    const int TB = 256;
    dim3 blk(TB);
    auto blocks = [](long long total, int tb) { return dim3((unsigned)((total + tb - 1) / tb)); };

    // Degree / normalization
    zero_f32<<<blocks(N, TB), blk, 0, stream>>>(deg, N);
    deg_edges<<<blocks(E, TB), blk, 0, stream>>>(dst, deg, E);
    make_dinv<<<blocks(N, TB), blk, 0, stream>>>(deg, dinv, N);
    make_norm_e<<<blocks(E, TB), blk, 0, stream>>>(src, dst, dinv, norme, E);

    const dim3 gemm_grid((unsigned)((N + 127) / 128));
    const dim3 node_grid = blocks((long long)N * FEAT, TB);
    const dim3 edge_grid = blocks((long long)E * 32, TB);

    // Layer 1: bufA = X@W1 ; bufB = agg(bufA)
    gcn_gemm_wmma<<<gemm_grid, blk, 0, stream>>>(x, W1, bufA, N, 0);
    agg_init<<<node_grid, blk, 0, stream>>>(bufA, dinv, b1, bufB, N);
    edge_scatter<<<edge_grid, blk, 0, stream>>>(bufA, src, dst, norme, bufB, E);

    // Layer 2: bufA = relu(bufB)@W2 ; bufB = agg(bufA)
    gcn_gemm_wmma<<<gemm_grid, blk, 0, stream>>>(bufB, W2, bufA, N, 1);
    agg_init<<<node_grid, blk, 0, stream>>>(bufA, dinv, b2, bufB, N);
    edge_scatter<<<edge_grid, blk, 0, stream>>>(bufA, src, dst, norme, bufB, E);

    // Layer 3 (no relu after): bufA = relu(bufB)@W3 ; bufB = agg(bufA)
    gcn_gemm_wmma<<<gemm_grid, blk, 0, stream>>>(bufB, W3, bufA, N, 1);
    agg_init<<<node_grid, blk, 0, stream>>>(bufA, dinv, b3, bufB, N);
    edge_scatter<<<edge_grid, blk, 0, stream>>>(bufA, src, dst, norme, bufB, E);

    // Mean pool + head
    zero_f32<<<blocks(NGRAPH * FEAT + NGRAPH, TB), blk, 0, stream>>>(sums, NGRAPH * FEAT + NGRAPH);
    pool_sum<<<node_grid, blk, 0, stream>>>(bufB, bat, sums, N);
    pool_cnt<<<blocks(N, TB), blk, 0, stream>>>(bat, counts, N);
    head_linear<<<blocks(NGRAPH * OUTF, TB), blk, 0, stream>>>(sums, counts, Wlin, blin, out);
}
